// SimplexConvLayer_39109972197652
// MI455X (gfx1250) — matrix operand order repaired and checked by reading, not compiled, not run on hardware
//
#include <hip/hip_runtime.h>
#include <cstdint>
#include <cstddef>

typedef __attribute__((ext_vector_type(16))) _Float16 v16h;
typedef __attribute__((ext_vector_type(8)))  _Float16 v8h;
typedef __attribute__((ext_vector_type(8)))  float    v8f;

#define DEV __device__ __forceinline__

DEV void lds_fence() { asm volatile("s_wait_dscnt 0" ::: "memory"); }
DEV float silu_f(float x) { return x / (1.0f + __expf(-x)); }

// ---------------------------------------------------------------------------
// WMMA tile GEMM: A is [16 x KT*32] f16 row-major in LDS (pitch in halfs),
// W is pre-swizzled f16 B-fragments: block (kt,nt) = 512 halfs, lane-major,
// 16 halfs per lane (layout: N = lane&15, k = (h&7) + 16*(h>>3) + 8*(lane>>4)).
// A fragment per ISA layout: row = lane&15, k = (h&7) + 16*(h>>3) + 8*(lane>>4)
// => two contiguous 16B LDS loads at (k0 + 8*(lane>>4)) and +16 halfs.
// ---------------------------------------------------------------------------
template <int KT>
DEV void wmma_gemm(const _Float16* __restrict__ A0, int pitch,
                   const _Float16* __restrict__ W, int lane, v8f acc[8]) {
#pragma unroll
  for (int nt = 0; nt < 8; ++nt)
#pragma unroll
    for (int i = 0; i < 8; ++i) acc[nt][i] = 0.0f;
  const int arow = lane & 15;
  const int koff = (lane >> 4) << 3;
#pragma unroll
  for (int kt = 0; kt < KT; ++kt) {
    const _Float16* ap = A0 + arow * pitch + kt * 32 + koff;
    v8h lo = *(const v8h*)ap;
    v8h hi = *(const v8h*)(ap + 16);
    v16h a = __builtin_shufflevector(lo, hi, 0, 1, 2, 3, 4, 5, 6, 7, 8, 9, 10,
                                     11, 12, 13, 14, 15);
#pragma unroll
    for (int nt = 0; nt < 8; ++nt) {
      v16h b = *(const v16h*)(W + ((size_t)(kt * 8 + nt)) * 512 +
                              (size_t)lane * 16);
      acc[nt] = __builtin_amdgcn_wmma_f32_16x16x32_f16(
          false, a, false, b, (short)0, acc[nt], false, false);
    }
  }
}

// D layout: VGPR r of lane l holds (M = r + 8*(l>>4), N = l&15).
template <bool DO_SILU>
DEV void epi_f16(const v8f acc[8], const float* __restrict__ bias,
                 _Float16* __restrict__ dst, int pitch, int lane) {
  const int nlo = lane & 15;
  const int rbase = (lane >> 4) << 3;
#pragma unroll
  for (int nt = 0; nt < 8; ++nt) {
    float bv = bias[nt * 16 + nlo];
#pragma unroll
    for (int r = 0; r < 8; ++r) {
      float x = acc[nt][r] + bv;
      if (DO_SILU) x = silu_f(x);
      dst[(rbase + r) * pitch + nt * 16 + nlo] = (_Float16)x;
    }
  }
}

DEV void epi_f32(const v8f acc[8], const float* __restrict__ bias,
                 float* __restrict__ dst, int pitch, int lane) {
  const int nlo = lane & 15;
  const int rbase = (lane >> 4) << 3;
#pragma unroll
  for (int nt = 0; nt < 8; ++nt) {
    float bv = bias[nt * 16 + nlo];
#pragma unroll
    for (int r = 0; r < 8; ++r)
      dst[(rbase + r) * pitch + nt * 16 + nlo] = acc[nt][r] + bv;
  }
}

// Scatter-add epilogue: silu(acc+bias) -> atomicAdd agg[node, :].
// idx_src: this lane's node index for edge-row (lane&15); row M fetched by shfl.
DEV void epi_atomic(const v8f acc[8], const float* __restrict__ bias,
                    float* __restrict__ agg, int idx_src, int lane) {
  const int nlo = lane & 15;
  const int rbase = (lane >> 4) << 3;
#pragma unroll
  for (int nt = 0; nt < 8; ++nt) {
    float bv = bias[nt * 16 + nlo];
#pragma unroll
    for (int r = 0; r < 8; ++r) {
      float x = silu_f(acc[nt][r] + bv);
      int node = __shfl(idx_src, rbase + r, 32);
      atomicAdd(agg + (size_t)node * 128 + nt * 16 + nlo, x);
    }
  }
}

// ---------------------------------------------------------------------------
// Weight packing: f32 (K x 128) row-major -> f16 B-fragment blocks.
// mode 0: identity rows. mode 1 (e2n_mw1, Ksrc=257): packed k<128 -> src 128+k
// (e_scal rows), 128..255 -> src k-128 (node rows), k==256 -> d2 row.
// Rows beyond source K are zero (lets activation pads stay garbage).
// ---------------------------------------------------------------------------
__global__ void pack_weight(const float* __restrict__ src,
                            _Float16* __restrict__ dst, int Ksrc, int ktiles,
                            int mode) {
  int idx = blockIdx.x * blockDim.x + threadIdx.x;
  int total = ktiles * 8 * 512;
  if (idx >= total) return;
  int h = idx & 15;
  int lane = (idx >> 4) & 31;
  int blk = idx >> 9;
  int nt = blk & 7;
  int kt = blk >> 3;
  int k = kt * 32 + ((lane >> 4) << 3) + (h & 7) + ((h >> 3) << 4);
  int n = nt * 16 + (lane & 15);
  int ks = k;
  bool ok;
  if (mode == 0) {
    ok = (k < Ksrc);
  } else {
    if (k < 128) { ks = 128 + k; ok = true; }
    else if (k < 256) { ks = k - 128; ok = true; }
    else if (k == 256) { ks = 256; ok = (Ksrc > 256); }
    else { ok = false; }
  }
  dst[(size_t)blk * 512 + lane * 16 + h] =
      ok ? (_Float16)src[(size_t)ks * 128 + n] : (_Float16)0.0f;
}

__global__ void zero_f32(float* __restrict__ p, long n) {
  long i = (long)blockIdx.x * blockDim.x + threadIdx.x;
  if (i < n) p[i] = 0.0f;
}

// ---------------------------------------------------------------------------
// Edge pass. One wave per 16-edge tile, 2 waves per block.
// Per-wave LDS: A [16][520] f16, B [16][136] f16, C [16][132] f32.
// bufA columns: X1 = [nf_row 0..127 | nf_col 128..255 | e_scal 256..383 |
//                     d2 384 | pad..415]; pm later at 384..511; X2 = cols
//                     256..511; X3 = [e_new 0..127 | nf_x 128..255 | d2 256].
// ---------------------------------------------------------------------------
#define PA 520
#define PB 136
#define PC 132
#define EDGE_WAVE_LDS (16 * PA * 2 + 16 * PB * 2 + 16 * PC * 4)  // 29440 B

__global__ __launch_bounds__(64) void edge_kernel(
    const float* __restrict__ nf, const float* __restrict__ ef,
    const float* __restrict__ pos, const int* __restrict__ eidx,
    const _Float16* __restrict__ w_m1, const float* __restrict__ b_m1,
    const _Float16* __restrict__ w_m2, const float* __restrict__ b_m2,
    const _Float16* __restrict__ w_u1, const float* __restrict__ b_u1,
    const _Float16* __restrict__ w_u2, const float* __restrict__ b_u2,
    const float* __restrict__ ln_g, const float* __restrict__ ln_b,
    const _Float16* __restrict__ w_em1, const float* __restrict__ b_em1,
    const _Float16* __restrict__ w_em2, const float* __restrict__ b_em2,
    float* __restrict__ agg, float* __restrict__ out_edge, int E_) {
  extern __shared__ char smem[];
  const int wave = threadIdx.x >> 5;
  const int lane = threadIdx.x & 31;
  char* wb = smem + (size_t)wave * EDGE_WAVE_LDS;
  _Float16* A = (_Float16*)wb;
  _Float16* Bh = (_Float16*)(wb + 16 * PA * 2);
  float* C = (float*)(wb + 16 * PA * 2 + 16 * PB * 2);

  const int tile = blockIdx.x * 2 + wave;
  if (tile * 16 >= E_) return;
  const int row = lane & 15;
  const int half = lane >> 4;
  const long e = (long)tile * 16 + row;
  const int er = eidx[e];
  const int ec = eidx[(long)E_ + e];

  float dx = pos[er * 3 + 0] - pos[ec * 3 + 0];
  float dy = pos[er * 3 + 1] - pos[ec * 3 + 1];
  float dz = pos[er * 3 + 2] - pos[ec * 3 + 2];
  float d2 = dx * dx + dy * dy + dz * dz;

  // Stage 1: build X1.
  {
    const float* pr = nf + (size_t)er * 128 + half * 64;
    const float* pc = nf + (size_t)ec * 128 + half * 64;
    const float* pe = ef + e * 131 + 3 + half * 64;
    _Float16* d0 = A + row * PA + half * 64;
    _Float16* d1 = A + row * PA + 128 + half * 64;
    _Float16* d2p = A + row * PA + 256 + half * 64;
    for (int i = 0; i < 64; ++i) d0[i] = (_Float16)pr[i];
    for (int i = 0; i < 64; ++i) d1[i] = (_Float16)pc[i];
    for (int i = 0; i < 64; ++i) d2p[i] = (_Float16)pe[i];
    if (half == 0) A[row * PA + 384] = (_Float16)d2;
  }
  lds_fence();

  v8f acc[8];
  // n2e msg MLP: h1 = silu(X1 @ mw1 + b1)  (K = 416 padded)
  wmma_gemm<13>(A, PA, w_m1, lane, acc);
  epi_f16<true>(acc, b_m1, Bh, PB, lane);
  lds_fence();
  // pm = silu(h1 @ mw2 + b2) -> bufA cols 384..511
  wmma_gemm<4>(Bh, PB, w_m2, lane, acc);
  epi_f16<true>(acc, b_m2, A + 384, PA, lane);
  lds_fence();
  // upd MLP: u = silu([e_scal|pm] @ uw1 + b1)
  wmma_gemm<8>(A + 256, PA, w_u1, lane, acc);
  epi_f16<true>(acc, b_u1, Bh, PB, lane);
  lds_fence();
  // upd = u @ uw2 + b2 (f32, no silu)
  wmma_gemm<4>(Bh, PB, w_u2, lane, acc);
  epi_f32(acc, b_u2, C, PC, lane);
  lds_fence();

  // LayerNorm(e_scal + upd); write edge output; stage e_new f16 at cols 0..127.
  {
    const float* es = ef + e * 131 + 3;
    float s = 0.f, s2 = 0.f;
    for (int i = 0; i < 64; ++i) {
      float v = es[half * 64 + i] + C[row * PC + half * 64 + i];
      s += v;
      s2 += v * v;
    }
    s += __shfl_xor(s, 16, 32);
    s2 += __shfl_xor(s2, 16, 32);
    float m = s * (1.0f / 128.0f);
    float var = s2 * (1.0f / 128.0f) - m * m;
    float rstd = rsqrtf(var + 1e-5f);
    float* oe = out_edge + e * 131;
    for (int i = 0; i < 64; ++i) {
      int c = half * 64 + i;
      float v = es[c] + C[row * PC + c];
      float y = (v - m) * rstd * ln_g[c] + ln_b[c];
      oe[3 + c] = y;
      A[row * PA + c] = (_Float16)y;
    }
    if (half == 0) {
      oe[0] = ef[e * 131 + 0];
      oe[1] = ef[e * 131 + 1];
      oe[2] = ef[e * 131 + 2];
      A[row * PA + 256] = (_Float16)d2;  // d2 column for X3
    }
  }
  lds_fence();

  // e2n message for row side: uses nf[col] (already at cols 128..255).
  wmma_gemm<9>(A, PA, w_em1, lane, acc);
  epi_f16<true>(acc, b_em1, Bh, PB, lane);
  lds_fence();
  wmma_gemm<4>(Bh, PB, w_em2, lane, acc);
  epi_atomic(acc, b_em2, agg, er, lane);

  // Restage nf[row] into cols 128..255 for the col side.
  {
    const float* pr = nf + (size_t)er * 128 + half * 64;
    _Float16* d1 = A + row * PA + 128 + half * 64;
    for (int i = 0; i < 64; ++i) d1[i] = (_Float16)pr[i];
  }
  lds_fence();
  wmma_gemm<9>(A, PA, w_em1, lane, acc);
  epi_f16<true>(acc, b_em1, Bh, PB, lane);
  lds_fence();
  wmma_gemm<4>(Bh, PB, w_em2, lane, acc);
  epi_atomic(acc, b_em2, agg, ec, lane);
}

// ---------------------------------------------------------------------------
// Node pass: nu = silu([nf|agg] @ uw1 + b1) @ uw2 + b2; out = LN(nf + nu).
// ---------------------------------------------------------------------------
#define PAN 264
#define NODE_WAVE_LDS (16 * PAN * 2 + 16 * PB * 2 + 16 * PC * 4)  // 21248 B

__global__ __launch_bounds__(64) void node_kernel(
    const float* __restrict__ nf, const float* __restrict__ agg,
    const _Float16* __restrict__ w_u1, const float* __restrict__ b_u1,
    const _Float16* __restrict__ w_u2, const float* __restrict__ b_u2,
    const float* __restrict__ ln_g, const float* __restrict__ ln_b,
    float* __restrict__ out_node, int ntiles) {
  extern __shared__ char smem[];
  const int wave = threadIdx.x >> 5;
  const int lane = threadIdx.x & 31;
  char* wb = smem + (size_t)wave * NODE_WAVE_LDS;
  _Float16* A = (_Float16*)wb;
  _Float16* Bh = (_Float16*)(wb + 16 * PAN * 2);
  float* C = (float*)(wb + 16 * PAN * 2 + 16 * PB * 2);

  const int tile = blockIdx.x * 2 + wave;
  if (tile >= ntiles) return;
  const int row = lane & 15;
  const int half = lane >> 4;
  const long n = (long)tile * 16 + row;

  {
    const float* p1 = nf + (size_t)n * 128 + half * 64;
    const float* p2 = agg + (size_t)n * 128 + half * 64;
    _Float16* d0 = A + row * PAN + half * 64;
    _Float16* d1 = A + row * PAN + 128 + half * 64;
    for (int i = 0; i < 64; ++i) d0[i] = (_Float16)p1[i];
    for (int i = 0; i < 64; ++i) d1[i] = (_Float16)p2[i];
  }
  lds_fence();

  v8f acc[8];
  wmma_gemm<8>(A, PAN, w_u1, lane, acc);
  epi_f16<true>(acc, b_u1, Bh, PB, lane);
  lds_fence();
  wmma_gemm<4>(Bh, PB, w_u2, lane, acc);
  epi_f32(acc, b_u2, C, PC, lane);
  lds_fence();

  {
    const float* xs = nf + (size_t)n * 128;
    float s = 0.f, s2 = 0.f;
    for (int i = 0; i < 64; ++i) {
      float v = xs[half * 64 + i] + C[row * PC + half * 64 + i];
      s += v;
      s2 += v * v;
    }
    s += __shfl_xor(s, 16, 32);
    s2 += __shfl_xor(s2, 16, 32);
    float m = s * (1.0f / 128.0f);
    float var = s2 * (1.0f / 128.0f) - m * m;
    float rstd = rsqrtf(var + 1e-5f);
    float* on = out_node + (size_t)n * 128;
    for (int i = 0; i < 64; ++i) {
      int c = half * 64 + i;
      float v = xs[c] + C[row * PC + c];
      on[c] = (v - m) * rstd * ln_g[c] + ln_b[c];
    }
  }
}

// ---------------------------------------------------------------------------
extern "C" void kernel_launch(void* const* d_in, const int* in_sizes, int n_in,
                              void* d_out, int out_size, void* d_ws,
                              size_t ws_size, hipStream_t stream) {
  const float* nf = (const float*)d_in[0];
  const float* ef = (const float*)d_in[1];
  const float* pos = (const float*)d_in[2];
  const int* eidx = (const int*)d_in[3];
  const float* n2e_mw1 = (const float*)d_in[4];
  const float* n2e_mb1 = (const float*)d_in[5];
  const float* n2e_mw2 = (const float*)d_in[6];
  const float* n2e_mb2 = (const float*)d_in[7];
  const float* n2e_uw1 = (const float*)d_in[8];
  const float* n2e_ub1 = (const float*)d_in[9];
  const float* n2e_uw2 = (const float*)d_in[10];
  const float* n2e_ub2 = (const float*)d_in[11];
  const float* n2e_g = (const float*)d_in[12];
  const float* n2e_b = (const float*)d_in[13];
  const float* e2n_mw1 = (const float*)d_in[14];
  const float* e2n_mb1 = (const float*)d_in[15];
  const float* e2n_mw2 = (const float*)d_in[16];
  const float* e2n_mb2 = (const float*)d_in[17];
  const float* e2n_uw1 = (const float*)d_in[18];
  const float* e2n_ub1 = (const float*)d_in[19];
  const float* e2n_uw2 = (const float*)d_in[20];
  const float* e2n_ub2 = (const float*)d_in[21];
  const float* e2n_g = (const float*)d_in[22];
  const float* e2n_b = (const float*)d_in[23];

  const int N_ = in_sizes[0] / 128;
  const int E_ = in_sizes[3] / 2;

  char* ws = (char*)d_ws;
  float* agg = (float*)ws;
  size_t off = (size_t)N_ * 128 * sizeof(float);
  auto take = [&](int ktiles) {
    _Float16* p = (_Float16*)(ws + off);
    off += (size_t)ktiles * 8 * 512 * sizeof(_Float16);
    return p;
  };
  _Float16* P_m1 = take(13);   // n2e_mw1 (385 -> 416)
  _Float16* P_m2 = take(4);    // n2e_mw2 (128)
  _Float16* P_u1 = take(8);    // n2e_uw1 (256)
  _Float16* P_u2 = take(4);    // n2e_uw2 (128)
  _Float16* P_em1 = take(9);   // e2n_mw1 (257 -> 288, remapped)
  _Float16* P_em2 = take(4);   // e2n_mw2 (128)
  _Float16* P_nu1 = take(8);   // e2n_uw1 (256)
  _Float16* P_nu2 = take(4);   // e2n_uw2 (128)

  auto packs = [&](const float* src, _Float16* dst, int K, int kt, int mode) {
    int total = kt * 8 * 512;
    pack_weight<<<(total + 255) / 256, 256, 0, stream>>>(src, dst, K, kt, mode);
  };
  packs(n2e_mw1, P_m1, 385, 13, 0);
  packs(n2e_mw2, P_m2, 128, 4, 0);
  packs(n2e_uw1, P_u1, 256, 8, 0);
  packs(n2e_uw2, P_u2, 128, 4, 0);
  packs(e2n_mw1, P_em1, 257, 9, 1);
  packs(e2n_mw2, P_em2, 128, 4, 0);
  packs(e2n_uw1, P_nu1, 256, 8, 0);
  packs(e2n_uw2, P_nu2, 128, 4, 0);

  long aggN = (long)N_ * 128;
  zero_f32<<<(int)((aggN + 255) / 256), 256, 0, stream>>>(agg, aggN);

  float* out_node = (float*)d_out;
  float* out_edge = out_node + (size_t)N_ * 128;

  int etiles = E_ / 16;
  int eblocks = (etiles + 1) / 2;
  edge_kernel<<<eblocks, 64, 2 * EDGE_WAVE_LDS, stream>>>(
      nf, ef, pos, eidx, P_m1, n2e_mb1, P_m2, n2e_mb2, P_u1, n2e_ub1, P_u2,
      n2e_ub2, n2e_g, n2e_b, P_em1, e2n_mb1, P_em2, e2n_mb2, agg, out_edge,
      E_);

  int ntiles = (N_ + 15) / 16;
  int nblocks = (ntiles + 1) / 2;
  node_kernel<<<nblocks, 64, 2 * NODE_WAVE_LDS, stream>>>(
      nf, agg, P_nu1, e2n_ub1, P_nu2, e2n_ub2, e2n_g, e2n_b, out_node, ntiles);
}
